// ENU_Light_GRU_9208409882934
// MI455X (gfx1250) — matrix operand order repaired and checked by reading, not compiled, not run on hardware
//
#include <hip/hip_runtime.h>

typedef __attribute__((ext_vector_type(16))) _Float16 v16h;
typedef __attribute__((ext_vector_type(8)))  _Float16 v8h;
typedef __attribute__((ext_vector_type(8)))  float    v8f;

#define T_DIM 256
#define B_DIM 64
#define I_DIM 512
#define H_DIM 1024
#define U_DIM 8
#define TB    (T_DIM * B_DIM)     // 16384
#define UH    (U_DIM * H_DIM)     // 8192
#define BH    (B_DIM * H_DIM)     // 65536
#define EPSV  1e-5f

// ---------------------------------------------------------------------------
// WMMA fragment loaders, contiguous k order shared by A and B (not the ISA table order)
// A: 16x32 f16.  lane L: M = L%16; halves e=0..15 -> K = k0 + (L/16)*16 + e
// B: 32x16 f16 (col n = row n of K-major weight). lane L: N = L%16,
//                 halves e=0..15 -> K = k0 + (L/16)*16 + e  (one 32B load)
// C/D: v8f, element i -> M = m0 + i + (L/16)*8, N = n0 + L%16
// ---------------------------------------------------------------------------
__device__ __forceinline__ v16h load_a_frag(const _Float16* __restrict__ A,
                                            int lda, int m0, int k0, int lid) {
  int row  = m0 + (lid & 15);
  int koff = k0 + ((lid >> 4) << 4);
  const _Float16* p = A + (size_t)row * lda + koff;
  v8h lo = *(const v8h*)p;
  v8h hi = *(const v8h*)(p + 8);
  return __builtin_shufflevector(lo, hi, 0, 1, 2, 3, 4, 5, 6, 7,
                                         8, 9, 10, 11, 12, 13, 14, 15);
}

__device__ __forceinline__ v16h load_b_frag(const _Float16* __restrict__ W,
                                            int ldw, int n0, int k0, int lid) {
  int col  = n0 + (lid & 15);
  int koff = k0 + ((lid >> 4) << 4);
  return *(const v16h*)(W + (size_t)col * ldw + koff);
}

#define WMMA_F16(A, B, C) \
  __builtin_amdgcn_wmma_f32_16x16x32_f16(false, (A), false, (B), (short)0, (C), false, false)

// ---------------------------------------------------------------------------
// Utility kernels
// ---------------------------------------------------------------------------
__global__ void cvt_f32_f16_kernel(const float* __restrict__ s,
                                   _Float16* __restrict__ d, int n) {
  int i = blockIdx.x * blockDim.x + threadIdx.x;
  if (i < n) d[i] = (_Float16)s[i];
}

__global__ void zero_h_kernel(float* __restrict__ h0f32,
                              _Float16* __restrict__ h16, int n) {
  int i = blockIdx.x * blockDim.x + threadIdx.x;
  if (i < n) { h0f32[i] = 0.f; h16[i] = (_Float16)0.f; }
}

// ---------------------------------------------------------------------------
// GEMM 1: zx_raw[TB, H] = xh[TB, I] @ Wzxh[H, I]^T   (f16 in, f32 out)
// Block = 4 waves (128 thr). Wave w: 16 rows (m0), 64 cols (n0..n0+63).
// ---------------------------------------------------------------------------
__global__ __launch_bounds__(128) void gemm_zx_kernel(
    const _Float16* __restrict__ xh, const _Float16* __restrict__ Wzxh,
    float* __restrict__ zx_raw) {
  int wave = threadIdx.x >> 5, lid = threadIdx.x & 31;
  int n0 = blockIdx.x * 64;
  int m0 = (blockIdx.y * 4 + wave) * 16;

  v8f acc[4] = {{}, {}, {}, {}};
  for (int k0 = 0; k0 < I_DIM; k0 += 32) {
    v16h a = load_a_frag(xh, I_DIM, m0, k0, lid);
#pragma unroll
    for (int j = 0; j < 4; ++j) {
      v16h b = load_b_frag(Wzxh, I_DIM, n0 + j * 16, k0, lid);
      acc[j] = WMMA_F16(a, b, acc[j]);
    }
  }
  int mbase = m0 + ((lid >> 4) << 3);
  int ncol  = n0 + (lid & 15);
#pragma unroll
  for (int j = 0; j < 4; ++j)
#pragma unroll
    for (int i = 0; i < 8; ++i)
      zx_raw[(size_t)(mbase + i) * H_DIM + ncol + j * 16] = acc[j][i];
}

// ---------------------------------------------------------------------------
// GEMM 2: hx = xh @ Whxh^T + b_unit; s_raw[m, h] = sum_u relu(hx[m, h*8+u]) * Wm
// Fused epilogue: bias + relu + per-8-column masked sum via shfl_xor.
// ---------------------------------------------------------------------------
__global__ __launch_bounds__(128) void gemm_hx_kernel(
    const _Float16* __restrict__ xh, const _Float16* __restrict__ Whxh,
    const float* __restrict__ b_unit, const float* __restrict__ Wh_in,
    const float* __restrict__ Wh_mask, float* __restrict__ s_raw) {
  int wave = threadIdx.x >> 5, lid = threadIdx.x & 31;
  int n0 = blockIdx.x * 64;            // column in UH space
  int m0 = (blockIdx.y * 4 + wave) * 16;

  v8f acc[4] = {{}, {}, {}, {}};
  for (int k0 = 0; k0 < I_DIM; k0 += 32) {
    v16h a = load_a_frag(xh, I_DIM, m0, k0, lid);
#pragma unroll
    for (int j = 0; j < 4; ++j) {
      v16h b = load_b_frag(Whxh, I_DIM, n0 + j * 16, k0, lid);
      acc[j] = WMMA_F16(a, b, acc[j]);
    }
  }
  int mbase = m0 + ((lid >> 4) << 3);
#pragma unroll
  for (int j = 0; j < 4; ++j) {
    int n = n0 + j * 16 + (lid & 15);  // column in [0, UH)
    int h = n >> 3;                    // output hidden index
    float w  = Wh_in[(size_t)h * UH + n] * Wh_mask[(size_t)h * UH + n];
    float bu = b_unit[n];
#pragma unroll
    for (int i = 0; i < 8; ++i) {
      float r = acc[j][i] + bu;
      r = r > 0.f ? r : 0.f;           // relu
      r *= w;                          // block-diag weight (== 1 at init)
      r += __shfl_xor(r, 1, 32);
      r += __shfl_xor(r, 2, 32);
      r += __shfl_xor(r, 4, 32);       // sum over group of 8 columns
      if ((lid & 7) == 0)
        s_raw[(size_t)(mbase + i) * H_DIM + h] = r;
    }
  }
}

// ---------------------------------------------------------------------------
// In-place training-mode BatchNorm over the batch dim, per timestep.
// y[t, b, h] -> (y - mu_th) * rsqrt(var_th + eps) * g[h] + beta[h]
// ---------------------------------------------------------------------------
__global__ __launch_bounds__(256) void bn_kernel(float* __restrict__ y,
                                                 const float* __restrict__ g,
                                                 const float* __restrict__ beta) {
  int t = blockIdx.x;
  float* base = y + (size_t)t * BH;
  for (int h = threadIdx.x; h < H_DIM; h += blockDim.x) {
    float s = 0.f, ss = 0.f;
    for (int b = 0; b < B_DIM; ++b) {
      float v = base[(size_t)b * H_DIM + h];
      s += v; ss += v * v;
    }
    float mu  = s * (1.f / B_DIM);
    float var = ss * (1.f / B_DIM) - mu * mu;
    float scale = rsqrtf(var + EPSV) * g[h];
    float shift = beta[h] - mu * scale;
    for (int b = 0; b < B_DIM; ++b) {
      size_t idx = (size_t)b * H_DIM + h;
      base[idx] = base[idx] * scale + shift;
    }
  }
}

// ---------------------------------------------------------------------------
// One recurrent step:
//   z  = sigmoid(a_t + h@Uz^T);  hu = relu(c_t + h@Uh^T)
//   h' = z*h + (1-z)*hu
// Block = 4 waves; wave: 16x16 tile of both GEMMs sharing the A fragment.
// Grid = (H/64, B/16).
// ---------------------------------------------------------------------------
__global__ __launch_bounds__(128) void scan_step_kernel(
    const _Float16* __restrict__ h16_prev, const float* __restrict__ hprev_f32,
    const _Float16* __restrict__ Uzh, const _Float16* __restrict__ Uhh,
    const float* __restrict__ a_t, const float* __restrict__ c_t,
    float* __restrict__ h_out_f32, _Float16* __restrict__ h16_next) {
  int wave = threadIdx.x >> 5, lid = threadIdx.x & 31;
  int n0 = blockIdx.x * 64 + wave * 16;
  int m0 = blockIdx.y * 16;

  v8f accZ = {}, accH = {};
  for (int k0 = 0; k0 < H_DIM; k0 += 32) {
    v16h a  = load_a_frag(h16_prev, H_DIM, m0, k0, lid);
    v16h bz = load_b_frag(Uzh, H_DIM, n0, k0, lid);
    v16h bh = load_b_frag(Uhh, H_DIM, n0, k0, lid);
    accZ = WMMA_F16(a, bz, accZ);
    accH = WMMA_F16(a, bh, accH);
  }
  int mbase = m0 + ((lid >> 4) << 3);
  int n = n0 + (lid & 15);
#pragma unroll
  for (int i = 0; i < 8; ++i) {
    size_t idx = (size_t)(mbase + i) * H_DIM + n;
    float z  = 1.f / (1.f + __expf(-(accZ[i] + a_t[idx])));
    float hu = accH[i] + c_t[idx];
    hu = hu > 0.f ? hu : 0.f;
    float hp = hprev_f32[idx];
    float hn = z * hp + (1.f - z) * hu;
    h_out_f32[idx] = hn;
    h16_next[idx]  = (_Float16)hn;
  }
}

// ---------------------------------------------------------------------------
extern "C" void kernel_launch(void* const* d_in, const int* in_sizes, int n_in,
                              void* d_out, int out_size, void* d_ws, size_t ws_size,
                              hipStream_t stream) {
  const float* x      = (const float*)d_in[0];
  const float* Wzx    = (const float*)d_in[1];
  const float* Whx    = (const float*)d_in[2];
  const float* Uz     = (const float*)d_in[3];
  const float* Uh     = (const float*)d_in[4];
  const float* b_unit = (const float*)d_in[5];
  const float* Wh_in  = (const float*)d_in[6];
  const float* gz     = (const float*)d_in[7];
  const float* bz     = (const float*)d_in[8];
  const float* gh     = (const float*)d_in[9];
  const float* bh     = (const float*)d_in[10];
  const float* Whmask = (const float*)d_in[11];
  float* out = (float*)d_out;

  // ---- workspace carve-up (~166 MB) ----
  char* ws = (char*)d_ws;
  _Float16* xh   = (_Float16*)ws; ws += (size_t)TB * I_DIM * 2;     // 16.8 MB
  _Float16* Wzxh = (_Float16*)ws; ws += (size_t)H_DIM * I_DIM * 2;  // 1 MB
  _Float16* Whxh = (_Float16*)ws; ws += (size_t)UH * I_DIM * 2;     // 8.4 MB
  _Float16* Uzh  = (_Float16*)ws; ws += (size_t)H_DIM * H_DIM * 2;  // 2 MB
  _Float16* Uhh  = (_Float16*)ws; ws += (size_t)H_DIM * H_DIM * 2;  // 2 MB
  float* zx_raw  = (float*)ws;    ws += (size_t)TB * H_DIM * 4;     // 67 MB
  float* s_raw   = (float*)ws;    ws += (size_t)TB * H_DIM * 4;     // 67 MB
  _Float16* h16a = (_Float16*)ws; ws += (size_t)BH * 2;
  _Float16* h16b = (_Float16*)ws; ws += (size_t)BH * 2;
  float* h0f32   = (float*)ws;    ws += (size_t)BH * 4;

  auto cvt = [&](const float* s, _Float16* d, int n) {
    cvt_f32_f16_kernel<<<(n + 255) / 256, 256, 0, stream>>>(s, d, n);
  };

  // ---- 1. precision converts ----
  cvt(x,   xh,   TB * I_DIM);
  cvt(Wzx, Wzxh, H_DIM * I_DIM);
  cvt(Whx, Whxh, UH * I_DIM);
  cvt(Uz,  Uzh,  H_DIM * H_DIM);
  cvt(Uh,  Uhh,  H_DIM * H_DIM);
  zero_h_kernel<<<(BH + 255) / 256, 256, 0, stream>>>(h0f32, h16a, BH);

  // ---- 2. time-parallel input projections (WMMA) ----
  gemm_zx_kernel<<<dim3(H_DIM / 64, TB / 64), 128, 0, stream>>>(xh, Wzxh, zx_raw);
  gemm_hx_kernel<<<dim3(UH / 64, TB / 64), 128, 0, stream>>>(
      xh, Whxh, b_unit, Wh_in, Whmask, s_raw);

  // ---- 3. per-step batch-stat BatchNorm (in place) ----
  bn_kernel<<<T_DIM, 256, 0, stream>>>(zx_raw, gz, bz);
  bn_kernel<<<T_DIM, 256, 0, stream>>>(s_raw, gh, bh);

  // ---- 4. sequential recurrence, one kernel per step ----
  for (int t = 0; t < T_DIM; ++t) {
    const _Float16* hprev16 = (t & 1) ? h16b : h16a;
    _Float16*       hnext16 = (t & 1) ? h16a : h16b;
    const float*    hprev32 = (t == 0) ? h0f32 : (out + (size_t)(t - 1) * BH);
    scan_step_kernel<<<dim3(H_DIM / 64, B_DIM / 16), 128, 0, stream>>>(
        hprev16, hprev32, Uzh, Uhh,
        zx_raw + (size_t)t * BH, s_raw + (size_t)t * BH,
        out + (size_t)t * BH, hnext16);
  }

  // ---- 5. h_final = last timeseries slice ----
  hipMemcpyAsync(out + (size_t)T_DIM * BH, out + (size_t)(T_DIM - 1) * BH,
                 (size_t)BH * sizeof(float), hipMemcpyDeviceToDevice, stream);
}
